// SimpleCyberGNN_22196390986479
// MI455X (gfx1250) — matrix operand order repaired and checked by reading, not compiled, not run on hardware
//
#include <hip/hip_runtime.h>
#include <math.h>

// Problem constants (from reference)
#define N_NODES 50000
#define N_EDGES 800000
#define F_IN    64
#define HID     32
#define NH1     4
#define NB      64
#define NOUT    2
#define EPRIME  (N_EDGES + N_NODES)   // edges + self loops

typedef __attribute__((ext_vector_type(16))) _Float16 v16h;
typedef __attribute__((ext_vector_type(8)))  float    v8f;

__device__ __forceinline__ float leaky02(float x){ return x > 0.f ? x : 0.2f * x; }
__device__ __forceinline__ float eluf(float x){ return x > 0.f ? x : (expf(x) - 1.f); }

// Monotone-bits float atomic max (lowers to global_atomic_max_i32 / min_u32)
__device__ __forceinline__ void atomicMaxFloat(float* addr, float val){
  if (val >= 0.f) atomicMax((int*)addr, __float_as_int(val));
  else            atomicMin((unsigned int*)addr, __float_as_uint(val));
}

// ---------------------------------------------------------------- fill
__global__ void fill_kernel(float* __restrict__ p, float v, int n){
  int i = blockIdx.x * blockDim.x + threadIdx.x;
  if (i < n) p[i] = v;
}

// ---------------------------------------------------------------- WMMA GEMM
// C[M,NCOL] = A[M,K] @ B[K,NCOL], fp32 in/out, f16 WMMA operands, f32 accum.
// One wave per 16x16 output tile; K/NCOL compile-time so the k-loop fully
// unrolls into back-to-back v_wmma with clause-grouped b128 loads.
template<int K, int NCOL>
__global__ void gemm_wmma_f16(const float* __restrict__ A, const float* __restrict__ Bw,
                              float* __restrict__ C, int M){
  int wave = (blockIdx.x * blockDim.x + threadIdx.x) >> 5;
  int lane = threadIdx.x & 31;
  constexpr int NTN = NCOL >> 4;
  int tm = wave / NTN;
  int tn = wave - tm * NTN;
  if (tm * 16 >= M) return;               // wave-uniform guard; EXEC all-1s inside
  int mrow = tm * 16 + (lane & 15);
  int ncol = tn * 16 + (lane & 15);
  int hi   = lane >> 4;                   // 0: lanes 0-15, 1: lanes 16-31
  v8f acc = {};
#pragma unroll
  for (int k0 = 0; k0 < K; k0 += 32){
    v16h a, b;
    // A 16x32 f16 layout: lane holds row (lane&15);
    //   e<8 -> K = k0 + hi*8 + e ; e>=8 -> K = k0 + 16 + hi*8 + (e-8)
    const float* arow = A + (size_t)mrow * K + k0 + hi * 8;
    const float4 p0 = ((const float4*)arow)[0];
    const float4 p1 = ((const float4*)arow)[1];
    const float4 p2 = ((const float4*)(arow + 16))[0];
    const float4 p3 = ((const float4*)(arow + 16))[1];
    a[0]=(_Float16)p0.x; a[1]=(_Float16)p0.y; a[2]=(_Float16)p0.z; a[3]=(_Float16)p0.w;
    a[4]=(_Float16)p1.x; a[5]=(_Float16)p1.y; a[6]=(_Float16)p1.z; a[7]=(_Float16)p1.w;
    a[8]=(_Float16)p2.x; a[9]=(_Float16)p2.y; a[10]=(_Float16)p2.z; a[11]=(_Float16)p2.w;
    a[12]=(_Float16)p3.x; a[13]=(_Float16)p3.y; a[14]=(_Float16)p3.z; a[15]=(_Float16)p3.w;
    // B 32x16 f16 layout: lane holds column (lane&15); e<8 -> K = k0 + hi*8 + e ; e>=8 -> K = k0 + 16 + hi*8 + (e-8)
    const float* bcol = Bw + (size_t)(k0 + hi * 8) * NCOL + ncol;
#pragma unroll
    for (int e = 0; e < 16; ++e) b[e] = (_Float16)bcol[(size_t)(e + (e >= 8 ? 8 : 0)) * NCOL];
    acc = __builtin_amdgcn_wmma_f32_16x16x32_f16(false, a, false, b,
                                                 (short)0, acc, false, false);
  }
  // C/D layout: VGPR r -> row r + hi*8, column = lane&15
#pragma unroll
  for (int r = 0; r < 8; ++r)
    C[(size_t)(tm * 16 + hi * 8 + r) * NCOL + ncol] = acc[r];
}

// ---------------------------------------------------------------- attention dot products
// xw row-major [n, H*C]; thread i = node*H + h ; row offset = i*C (C == HID, 128B aligned)
__global__ void attn_dots(const float* __restrict__ xw, const float* __restrict__ att_s,
                          const float* __restrict__ att_d, float* __restrict__ as_out,
                          float* __restrict__ ad_out, int n, int H){
  int i = blockIdx.x * blockDim.x + threadIdx.x;
  if (i >= n * H) return;
  int h = i % H;
  const float4* row = (const float4*)(xw    + (size_t)i * HID);
  const float4* as  = (const float4*)(att_s + (size_t)h * HID);
  const float4* ad  = (const float4*)(att_d + (size_t)h * HID);
  float s = 0.f, d = 0.f;
#pragma unroll
  for (int c = 0; c < HID / 4; ++c){
    float4 v = row[c], a = as[c], b = ad[c];
    s += v.x*a.x + v.y*a.y + v.z*a.z + v.w*a.w;
    d += v.x*b.x + v.y*b.y + v.z*b.z + v.w*b.w;
  }
  as_out[i] = s;
  ad_out[i] = d;
}

// ---------------------------------------------------------------- segment-softmax pass 1: max
__global__ void edge_max(const int* __restrict__ ei, const float* __restrict__ as_,
                         const float* __restrict__ ad_, float* __restrict__ mmax, int H){
  int i = blockIdx.x * blockDim.x + threadIdx.x;
  if (i >= EPRIME * H) return;
  int e = i / H, h = i - e * H;
  int src, dst;
  if (e < N_EDGES){ src = ei[e]; dst = ei[N_EDGES + e]; }
  else            { src = dst = e - N_EDGES; }
  float ev = leaky02(as_[src * H + h] + ad_[dst * H + h]);
  atomicMaxFloat(&mmax[dst * H + h], ev);
}

// ---------------------------------------------------------------- pass 2: exp + denom
__global__ void edge_expsum(const int* __restrict__ ei, const float* __restrict__ as_,
                            const float* __restrict__ ad_, const float* __restrict__ mmax,
                            float* __restrict__ denom, float* __restrict__ exbuf, int H){
  int i = blockIdx.x * blockDim.x + threadIdx.x;
  if (i >= EPRIME * H) return;
  int e = i / H, h = i - e * H;
  int src, dst;
  if (e < N_EDGES){ src = ei[e]; dst = ei[N_EDGES + e]; }
  else            { src = dst = e - N_EDGES; }
  float ev = leaky02(as_[src * H + h] + ad_[dst * H + h]);
  float ex = expf(ev - mmax[dst * H + h]);
  exbuf[i] = ex;
  atomicAdd(&denom[dst * H + h], ex);
}

// ---------------------------------------------------------------- pass 3: weighted scatter-add
// One wave per (edge, head); lane == channel (HID == 32 == wave32 width).
// Edge metadata forced wave-uniform into SGPRs; gather + atomic fully coalesced.
__global__ void edge_agg(const int* __restrict__ ei, const float* __restrict__ exbuf,
                         const float* __restrict__ denom, const float* __restrict__ xw,
                         float* __restrict__ agg, int H){
  int w    = (blockIdx.x * blockDim.x + threadIdx.x) >> 5;   // (edge, head)
  int lane = threadIdx.x & 31;                               // channel
  if (w >= EPRIME * H) return;                               // wave-uniform guard
  int e = w / H, h = w - e * H;
  int src, dst;
  if (e < N_EDGES){ src = ei[e]; dst = ei[N_EDGES + e]; }
  else            { src = dst = e - N_EDGES; }
  src = __builtin_amdgcn_readfirstlane(src);
  dst = __builtin_amdgcn_readfirstlane(dst);
  float alpha = exbuf[w] / denom[dst * H + h];
  int HC = H * HID;
  float v = alpha * xw[(size_t)src * HC + h * HID + lane];
  atomicAdd(&agg[(size_t)dst * HC + h * HID + lane], v);
}

// ---------------------------------------------------------------- bias + ELU (in place)
__global__ void bias_elu(float* __restrict__ a, const float* __restrict__ bias, int n, int W){
  int i = blockIdx.x * blockDim.x + threadIdx.x;
  if (i >= n * W) return;
  a[i] = eluf(a[i] + bias[i % W]);
}

// ---------------------------------------------------------------- global mean pool
__global__ void pool_sum(const float* __restrict__ h2, const int* __restrict__ batch,
                         float* __restrict__ sums){
  int i = blockIdx.x * blockDim.x + threadIdx.x;
  if (i >= N_NODES * HID) return;
  int n = i / HID, c = i - n * HID;
  atomicAdd(&sums[batch[n] * HID + c], h2[i]);
}
__global__ void pool_cnt(const int* __restrict__ batch, float* __restrict__ cnts){
  int n = blockIdx.x * blockDim.x + threadIdx.x;
  if (n >= N_NODES) return;
  atomicAdd(&cnts[batch[n]], 1.f);
}

// ---------------------------------------------------------------- classifier MLP + log_softmax
__global__ void classifier(const float* __restrict__ sums, const float* __restrict__ cnts,
                           const float* __restrict__ Wc1, const float* __restrict__ bc1,
                           const float* __restrict__ Wc2, const float* __restrict__ bc2,
                           float* __restrict__ out){
  int g = threadIdx.x;
  if (g >= NB) return;
  float inv = 1.f / fmaxf(cnts[g], 1.f);
  float pooled[HID];
#pragma unroll
  for (int c = 0; c < HID; ++c) pooled[c] = sums[g * HID + c] * inv;
  float t[16];
#pragma unroll
  for (int j = 0; j < 16; ++j){
    float s = bc1[j];
    for (int c = 0; c < HID; ++c) s += pooled[c] * Wc1[c * 16 + j];
    t[j] = fmaxf(s, 0.f);
  }
  float l[NOUT];
#pragma unroll
  for (int o = 0; o < NOUT; ++o){
    float s = bc2[o];
    for (int j = 0; j < 16; ++j) s += t[j] * Wc2[j * NOUT + o];
    l[o] = s;
  }
  float m = fmaxf(l[0], l[1]);
  float lse = m + logf(expf(l[0] - m) + expf(l[1] - m));
  out[g * NOUT + 0] = l[0] - lse;
  out[g * NOUT + 1] = l[1] - lse;
}

static inline int ceil_div(long long a, int b){ return (int)((a + b - 1) / b); }

extern "C" void kernel_launch(void* const* d_in, const int* in_sizes, int n_in,
                              void* d_out, int out_size, void* d_ws, size_t ws_size,
                              hipStream_t stream){
  (void)in_sizes; (void)n_in; (void)out_size; (void)ws_size;
  const float* x    = (const float*)d_in[0];
  const int*   ei   = (const int*)  d_in[1];
  const int*   batch= (const int*)  d_in[2];
  const float* W1   = (const float*)d_in[3];
  const float* at_s1= (const float*)d_in[4];
  const float* at_d1= (const float*)d_in[5];
  const float* b1   = (const float*)d_in[6];
  const float* W2   = (const float*)d_in[7];
  const float* at_s2= (const float*)d_in[8];
  const float* at_d2= (const float*)d_in[9];
  const float* b2   = (const float*)d_in[10];
  const float* Wc1  = (const float*)d_in[11];
  const float* bc1  = (const float*)d_in[12];
  const float* Wc2  = (const float*)d_in[13];
  const float* bc2  = (const float*)d_in[14];
  float* out = (float*)d_out;

  // workspace carve-up (floats)
  float* ws = (float*)d_ws;
  size_t off = 0;
  float* xw1 = ws + off; off += (size_t)N_NODES * NH1 * HID;   // [N,128]
  float* as1 = ws + off; off += (size_t)N_NODES * NH1;
  float* ad1 = ws + off; off += (size_t)N_NODES * NH1;
  float* m1  = ws + off; off += (size_t)N_NODES * NH1;
  float* dn1 = ws + off; off += (size_t)N_NODES * NH1;
  float* ex1 = ws + off; off += (size_t)EPRIME  * NH1;
  float* ag1 = ws + off; off += (size_t)N_NODES * NH1 * HID;   // becomes h1 after bias_elu
  float* xw2 = ws + off; off += (size_t)N_NODES * HID;
  float* as2 = ws + off; off += (size_t)N_NODES;
  float* ad2 = ws + off; off += (size_t)N_NODES;
  float* m2  = ws + off; off += (size_t)N_NODES;
  float* dn2 = ws + off; off += (size_t)N_NODES;
  float* ex2 = ws + off; off += (size_t)EPRIME;
  float* ag2 = ws + off; off += (size_t)N_NODES * HID;         // becomes h2
  float* sums= ws + off; off += (size_t)NB * HID;
  float* cnts= ws + off; off += (size_t)NB;

  const int TB = 256;
  const float NEG_INF = -1e30f;

  // ---- init accumulators
  fill_kernel<<<ceil_div((long long)N_NODES*NH1, TB), TB, 0, stream>>>(m1, NEG_INF, N_NODES*NH1);
  fill_kernel<<<ceil_div((long long)N_NODES*NH1, TB), TB, 0, stream>>>(dn1, 0.f, N_NODES*NH1);
  fill_kernel<<<ceil_div((long long)N_NODES*NH1*HID, TB), TB, 0, stream>>>(ag1, 0.f, N_NODES*NH1*HID);
  fill_kernel<<<ceil_div(N_NODES, TB), TB, 0, stream>>>(m2, NEG_INF, N_NODES);
  fill_kernel<<<ceil_div(N_NODES, TB), TB, 0, stream>>>(dn2, 0.f, N_NODES);
  fill_kernel<<<ceil_div((long long)N_NODES*HID, TB), TB, 0, stream>>>(ag2, 0.f, N_NODES*HID);
  fill_kernel<<<ceil_div(NB*HID, TB), TB, 0, stream>>>(sums, 0.f, NB*HID);
  fill_kernel<<<ceil_div(NB, TB), TB, 0, stream>>>(cnts, 0.f, NB);

  // ---- layer 1: xw1 = x @ W1  (WMMA, K=64 fully unrolled -> 2 back-to-back v_wmma)
  {
    long long waves = (long long)(N_NODES/16) * ((NH1*HID)/16);
    gemm_wmma_f16<F_IN, NH1*HID><<<ceil_div(waves*32, 128), 128, 0, stream>>>(x, W1, xw1, N_NODES);
  }
  attn_dots<<<ceil_div((long long)N_NODES*NH1, TB), TB, 0, stream>>>(xw1, at_s1, at_d1, as1, ad1, N_NODES, NH1);
  edge_max   <<<ceil_div((long long)EPRIME*NH1, TB), TB, 0, stream>>>(ei, as1, ad1, m1, NH1);
  edge_expsum<<<ceil_div((long long)EPRIME*NH1, TB), TB, 0, stream>>>(ei, as1, ad1, m1, dn1, ex1, NH1);
  edge_agg   <<<ceil_div((long long)EPRIME*NH1*32, TB), TB, 0, stream>>>(ei, ex1, dn1, xw1, ag1, NH1);
  bias_elu   <<<ceil_div((long long)N_NODES*NH1*HID, TB), TB, 0, stream>>>(ag1, b1, N_NODES, NH1*HID);

  // ---- layer 2: xw2 = h1 @ W2 (WMMA, K=128 fully unrolled -> 4 back-to-back v_wmma)
  {
    long long waves = (long long)(N_NODES/16) * (HID/16);
    gemm_wmma_f16<NH1*HID, HID><<<ceil_div(waves*32, 128), 128, 0, stream>>>(ag1, W2, xw2, N_NODES);
  }
  attn_dots<<<ceil_div(N_NODES, TB), TB, 0, stream>>>(xw2, at_s2, at_d2, as2, ad2, N_NODES, 1);
  edge_max   <<<ceil_div(EPRIME, TB), TB, 0, stream>>>(ei, as2, ad2, m2, 1);
  edge_expsum<<<ceil_div(EPRIME, TB), TB, 0, stream>>>(ei, as2, ad2, m2, dn2, ex2, 1);
  edge_agg   <<<ceil_div((long long)EPRIME*32, TB), TB, 0, stream>>>(ei, ex2, dn2, xw2, ag2, 1);
  bias_elu   <<<ceil_div((long long)N_NODES*HID, TB), TB, 0, stream>>>(ag2, b2, N_NODES, HID);

  // ---- global mean pool + classifier
  pool_sum<<<ceil_div((long long)N_NODES*HID, TB), TB, 0, stream>>>(ag2, batch, sums);
  pool_cnt<<<ceil_div(N_NODES, TB), TB, 0, stream>>>(batch, cnts);
  classifier<<<1, NB, 0, stream>>>(sums, cnts, Wc1, bc1, Wc2, bc2, out);
}
